// TrainGRPOModuleJax_19653770346605
// MI455X (gfx1250) — matrix operand order repaired and checked by reading, not compiled, not run on hardware
//
#include <hip/hip_runtime.h>
#include <hip/hip_bf16.h>
#include <stdint.h>

#define VOCAB 32000
#define TPB 256
#define ROWB (VOCAB * 2) /* 64000 bytes per row */
#define T_OUT 1024
#define T_IN 1025
#define NB 4
#define EPS_LOW 0.2f
#define EPS_HIGH 0.3f

typedef float v8f __attribute__((ext_vector_type(8)));
typedef __bf16 v16bf __attribute__((ext_vector_type(16)));
typedef unsigned short v16us __attribute__((ext_vector_type(16)));

__device__ __forceinline__ float bf_lo(unsigned u) { return __uint_as_float(u << 16); }
__device__ __forceinline__ float bf_hi(unsigned u) { return __uint_as_float(u & 0xFFFF0000u); }

__device__ __forceinline__ unsigned short f2bf(float f) {
    unsigned u = __float_as_uint(f);
    unsigned r = u + 0x7FFFu + ((u >> 16) & 1u); // round-to-nearest-even
    return (unsigned short)(r >> 16);
}

// ---------------------------------------------------------------------------
// Kernel A: one workgroup per (b, t) token row.
// Stage the 64000-byte bf16 row into LDS with CDNA5 async copies, then do a
// max pass and an exp-sum pass out of LDS. Emits per-token logp (to d_out),
// token entropy and per-token loss (to workspace).
// ---------------------------------------------------------------------------
__global__ __launch_bounds__(TPB) void grpo_row_kernel(
    const unsigned short* __restrict__ logits, // bf16 bits, (4, 1025, 32000)
    const int* __restrict__ input_ids,         // (4, 1025)
    const float* __restrict__ advantages,      // (4,)
    float* __restrict__ d_out,                 // [1 + 4096 + 4 + 4]
    float* __restrict__ ws_ent,                // (4096,)
    float* __restrict__ ws_ploss)              // (4096,)
{
    __shared__ __align__(16) unsigned char s_row[ROWB];
    __shared__ float s_red[TPB];

    const int tid = threadIdx.x;
    const int row = blockIdx.x; // 0..4095, b-major
    const int b = row >> 10;
    const int t = row & 1023;

    const unsigned short* rowg = logits + ((size_t)b * T_IN + (size_t)t) * (size_t)VOCAB;
    // Low 32 bits of the flat shared address are the LDS byte offset
    // (ISA 10.2, LDS aperture).
    const unsigned lds_base = (unsigned)(uintptr_t)&s_row[0];

    // Async copy HBM -> LDS: 16 iterations x 256 lanes x 16B (last partial:
    // only lanes with off < 64000 issue). Tracked by ASYNCcnt.
#pragma unroll
    for (int it = 0; it < 16; ++it) {
        int off = it * (TPB * 16) + tid * 16;
        if (off < ROWB) {
            unsigned laddr = lds_base + (unsigned)off;
            asm volatile("global_load_async_to_lds_b128 %0, %1, %2"
                         :: "v"(laddr), "v"(off), "s"(rowg)
                         : "memory");
        }
    }
    asm volatile("s_wait_asynccnt 0" ::: "memory");
    __syncthreads();

    const uint4* r4 = (const uint4*)s_row;

    // Pass 1: row max (8 bf16 per 16B LDS load; bf16->f32 is a shift).
    float lmax = -3.0e38f;
    for (int c = tid; c < VOCAB / 8; c += TPB) {
        uint4 d = r4[c];
        lmax = fmaxf(lmax, fmaxf(bf_lo(d.x), bf_hi(d.x)));
        lmax = fmaxf(lmax, fmaxf(bf_lo(d.y), bf_hi(d.y)));
        lmax = fmaxf(lmax, fmaxf(bf_lo(d.z), bf_hi(d.z)));
        lmax = fmaxf(lmax, fmaxf(bf_lo(d.w), bf_hi(d.w)));
    }
    s_red[tid] = lmax;
    __syncthreads();
    for (int s = TPB / 2; s > 0; s >>= 1) {
        if (tid < s) s_red[tid] = fmaxf(s_red[tid], s_red[tid + s]);
        __syncthreads();
    }
    const float m = s_red[0];
    __syncthreads();

    // Pass 2: S = sum exp(x-m), SX = sum exp(x-m)*x.
    float se = 0.f, sex = 0.f;
    for (int c = tid; c < VOCAB / 8; c += TPB) {
        uint4 d = r4[c];
        unsigned w[4] = {d.x, d.y, d.z, d.w};
#pragma unroll
        for (int j = 0; j < 4; ++j) {
            float x0 = bf_lo(w[j]);
            float x1 = bf_hi(w[j]);
            float e0 = __expf(x0 - m);
            float e1 = __expf(x1 - m);
            se += e0 + e1;
            sex = fmaf(e0, x0, sex);
            sex = fmaf(e1, x1, sex);
        }
    }
    s_red[tid] = se;
    __syncthreads();
    for (int s = TPB / 2; s > 0; s >>= 1) {
        if (tid < s) s_red[tid] += s_red[tid + s];
        __syncthreads();
    }
    const float S = s_red[0];
    __syncthreads();
    s_red[tid] = sex;
    __syncthreads();
    for (int s = TPB / 2; s > 0; s >>= 1) {
        if (tid < s) s_red[tid] += s_red[tid + s];
        __syncthreads();
    }
    const float SX = s_red[0];

    if (tid == 0) {
        int id = input_ids[(size_t)b * T_IN + (size_t)t + 1]; // chosen_ids = input_ids[:,1:]
        unsigned short cb = ((const unsigned short*)s_row)[id];
        float chosen = __uint_as_float(((unsigned)cb) << 16);
        float lse = m + __logf(S);
        float logp = chosen - lse;            // TEMPERATURE == 1.0
        float entropy = lse - SX / S;         // -sum p*(x - lse), sum p == 1
        float ratio = __expf(logp - logp);    // old = stop_grad(new) -> 1
        float clipped = fminf(fmaxf(ratio, 1.f - EPS_LOW), 1.f + EPS_HIGH);
        float adv = advantages[b];
        float ploss = -fminf(ratio * adv, clipped * adv);
        d_out[1 + row] = logp;
        ws_ent[row] = entropy;
        ws_ploss[row] = ploss;
    }
}

// ---------------------------------------------------------------------------
// Kernel B: one workgroup per batch. Masked time-axis sums. The masked
// entropy dot products run on the matrix unit: v_wmma_f32_16x16x32_bf16 with
// A = 16 rows of 32 entropies and B arranged so column n = mask[n*32+k];
// diag(D) gives the 16 partial dot products (512 elems/WMMA, 2 chunks).
// ---------------------------------------------------------------------------
__global__ __launch_bounds__(TPB) void grpo_batch_kernel(
    const float* __restrict__ ws_ent,
    const float* __restrict__ ws_ploss,
    const int* __restrict__ labels, // (4, 1025)
    float* __restrict__ d_out,
    float* __restrict__ ws_lossnum, // (4,)
    float* __restrict__ ws_masksum) // (4,)
{
    __shared__ float sent[T_OUT];
    __shared__ float smask[T_OUT];
    __shared__ float strunc[T_OUT];
    __shared__ float red[TPB];
    __shared__ float sh_masksum, sh_truncsum, sh_plosssum;

    const int tid = threadIdx.x;
    const int b = blockIdx.x;

    float pmask = 0.f, pl = 0.f;
    for (int t = tid; t < T_OUT; t += TPB) {
        float e = ws_ent[b * T_OUT + t];
        float mk = (float)labels[b * T_IN + 1 + t]; // mask_loss = labels[:,1:]
        sent[t] = e;
        smask[t] = mk;
        pmask += mk;
        pl += ws_ploss[b * T_OUT + t] * mk;
    }
    __syncthreads();

    // cumsum-based truncation mask: valid & 4 <= cum <= 100
    if (tid == 0) {
        int c = 0;
        for (int t = 0; t < T_OUT; ++t) {
            int v = (smask[t] == 1.f);
            c += v;
            strunc[t] = (v && c >= 4 && c <= 100) ? 1.f : 0.f;
        }
    }

    red[tid] = pmask;
    __syncthreads(); // also publishes strunc
    for (int s = TPB / 2; s > 0; s >>= 1) {
        if (tid < s) red[tid] += red[tid + s];
        __syncthreads();
    }
    if (tid == 0) sh_masksum = red[0];
    __syncthreads();

    red[tid] = pl;
    __syncthreads();
    for (int s = TPB / 2; s > 0; s >>= 1) {
        if (tid < s) red[tid] += red[tid + s];
        __syncthreads();
    }
    if (tid == 0) sh_plosssum = red[0];
    __syncthreads();

    float pt = 0.f;
    for (int t = tid; t < T_OUT; t += TPB) pt += strunc[t];
    red[tid] = pt;
    __syncthreads();
    for (int s = TPB / 2; s > 0; s >>= 1) {
        if (tid < s) red[tid] += red[tid + s];
        __syncthreads();
    }
    if (tid == 0) sh_truncsum = red[0];
    __syncthreads();

    if (tid < 32) { // wave 0, EXEC all-ones within the wave (WMMA requirement)
        const int lane = tid;
        const int r = lane & 15;
        const bool hiHalf = lane >= 16;
        v8f cm = {0.f, 0.f, 0.f, 0.f, 0.f, 0.f, 0.f, 0.f};
        v8f ct = {0.f, 0.f, 0.f, 0.f, 0.f, 0.f, 0.f, 0.f};
#pragma unroll
        for (int chunk = 0; chunk < 2; ++chunk) {
            const int base = chunk * 512;
            // A (16x32 bf16): lanes 0-15 hold A[r, 0..15];
            //                 lanes 16-31 hold A[r, 16..31].
            v16us ua, ubm, ubt;
            const int arow = base + r * 32 + (hiHalf ? 16 : 0);
#pragma unroll
            for (int j = 0; j < 8; ++j) {
                ua[j]     = f2bf(sent[arow + j]);
                ua[8 + j] = f2bf(sent[arow + 8 + j]);
            }
            // B (32x16 bf16): lane n<16 holds B[0..15, n], lane n+16 holds
            // B[16..31, n]; column n = mask[base + n*32 + k].
            const int bcol = base + r * 32 + (hiHalf ? 16 : 0);
#pragma unroll
            for (int k = 0; k < 16; ++k) {
                ubm[k] = f2bf(smask[bcol + k]);
                ubt[k] = f2bf(strunc[bcol + k]);
            }
            v16bf a  = __builtin_bit_cast(v16bf, ua);
            v16bf bm = __builtin_bit_cast(v16bf, ubm);
            v16bf bt = __builtin_bit_cast(v16bf, ubt);
            cm = __builtin_amdgcn_wmma_f32_16x16x32_bf16(false, a, false, bm,
                                                         (short)0, cm, false, false);
            ct = __builtin_amdgcn_wmma_f32_16x16x32_bf16(false, a, false, bt,
                                                         (short)0, ct, false, false);
        }
        // diag(D): m<8 -> VGPR m, lane m; m>=8 -> VGPR m-8, lane m+16.
        float dm = 0.f, dt = 0.f;
        if (lane < 8)        { dm = cm[lane];      dt = ct[lane]; }
        else if (lane >= 24) { dm = cm[lane - 24]; dt = ct[lane - 24]; }
#pragma unroll
        for (int off = 16; off > 0; off >>= 1) {
            dm += __shfl_xor(dm, off);
            dt += __shfl_xor(dt, off);
        }
        if (lane == 0) {
            d_out[1 + NB * T_OUT + b]      = dm / sh_masksum;  // avg_entropy
            d_out[1 + NB * T_OUT + NB + b] = dt / sh_truncsum; // avg_entropy_trunc
            ws_lossnum[b] = sh_plosssum;
            ws_masksum[b] = sh_masksum;
        }
    }
}

// ---------------------------------------------------------------------------
// Kernel C: scalar loss = sum_b lossnum_b / sum_b masksum_b
// ---------------------------------------------------------------------------
__global__ void grpo_final_kernel(const float* __restrict__ ws_lossnum,
                                  const float* __restrict__ ws_masksum,
                                  float* __restrict__ d_out)
{
    if (threadIdx.x == 0 && blockIdx.x == 0) {
        float n = 0.f, d = 0.f;
#pragma unroll
        for (int i = 0; i < NB; ++i) { n += ws_lossnum[i]; d += ws_masksum[i]; }
        d_out[0] = n / d;
    }
}

extern "C" void kernel_launch(void* const* d_in, const int* in_sizes, int n_in,
                              void* d_out, int out_size, void* d_ws, size_t ws_size,
                              hipStream_t stream)
{
    (void)in_sizes; (void)n_in; (void)out_size; (void)ws_size;
    const unsigned short* logits = (const unsigned short*)d_in[0]; // bf16 bits
    const int* input_ids = (const int*)d_in[1];
    const int* labels = (const int*)d_in[2];
    const float* advantages = (const float*)d_in[3];
    float* out = (float*)d_out;

    float* ws = (float*)d_ws;
    float* ws_ent     = ws;        // 4096 f32
    float* ws_ploss   = ws + 4096; // 4096 f32
    float* ws_lossnum = ws + 8192; // 4 f32
    float* ws_masksum = ws + 8196; // 4 f32

    grpo_row_kernel<<<NB * T_OUT, TPB, 0, stream>>>(logits, input_ids, advantages,
                                                    out, ws_ent, ws_ploss);
    grpo_batch_kernel<<<NB, TPB, 0, stream>>>(ws_ent, ws_ploss, labels,
                                              out, ws_lossnum, ws_masksum);
    grpo_final_kernel<<<1, 32, 0, stream>>>(ws_lossnum, ws_masksum, out);
}
